// RTDETRTransformerv2_73246372266055
// MI455X (gfx1250) — matrix operand order repaired and checked by reading, not compiled, not run on hardware
//
#include <hip/hip_runtime.h>
#include <cstddef>
#include <cstdint>

// ======================= problem constants =======================
constexpr int CB     = 16;            // batch
constexpr int CL     = 8400;          // memory length (80*80+40*40+20*20)
constexpr int CH     = 256;           // hidden
constexpr int CNH    = 8;             // heads
constexpr int CHD    = CH / CNH;      // 32
constexpr int CNLEV  = 3;
constexpr int CNPT   = 4;
constexpr int CNLAYER= 6;
constexpr int CNQ    = 300;
constexpr int CNC    = 80;
constexpr int CDFF   = 1024;

typedef __attribute__((ext_vector_type(16))) _Float16 v16h;
typedef __attribute__((ext_vector_type(8)))  _Float16 v8h;
typedef __attribute__((ext_vector_type(8)))  float    v8f;

union V16 { v16h v; v8h h[2]; };

// ======================= WMMA GEMM =======================
// C[M,N] = act(A[M,K] @ W[K,N] + bias), A row-major (lda=K), W row-major with
// row stride ldw (lets us slice fused-QKV weight columns), C compact row-major.
// Requirements: M % 64 == 0, K % 32 == 0, N % 4 == 0 (N guarded per float4).
// Block: 256 thr = 8 waves; block tile 64x64; wave tile 16x32 (2 WMMAs sharing
// the A fragment). LDS DOUBLE-BUFFERED: stage tile k+1 while computing tile k,
// one barrier per k-step, global loads overlap the WMMAs. A staged [m][k] as
// f16; B staged TRANSPOSED [n][k] so fragments read as contiguous b128 loads.
// Uniform fullN fast path keeps exec-mask churn out of the steady-state loop.
__global__ __launch_bounds__(256) void gemm_f16_wmma_kernel(
    const float* __restrict__ A, const float* __restrict__ W,
    const float* __restrict__ bias, float* __restrict__ C,
    int N, int K, int ldw, int relu)
{
  __shared__ _Float16 sA [2][64 * 32];   // [m][k]
  __shared__ _Float16 sBt[2][64 * 32];   // [n][k] (transposed)
  const int tid  = threadIdx.x;
  const int wave = tid >> 5;
  const int lane = tid & 31;
  const int m0 = blockIdx.y * 64;
  const int n0 = blockIdx.x * 64;
  const int wm = (wave >> 1) * 16;   // 0,16,32,48
  const int wn = (wave & 1) * 32;    // 0,32
  const int lrow = lane & 15;
  const int hsel = lane >> 4;        // lane half select
  const int kb   = hsel * 8;         // A-frag K split (ISA 16-bit A 16x32 layout)
  const bool fullN = (n0 + 64 <= N); // uniform: scalar branch in stage path

  // per-thread staging coordinates
  const int ar = tid >> 2;           // A row 0..63
  const int ac = (tid & 3) * 8;      // A col 0,8,16,24
  const int bk = tid >> 3;           // B k-row 0..31
  const int bc = (tid & 7) * 8;      // B col 0,8,..,56

  auto stage = [&](int bsel, int kk0) {
    // ---- A tile 64x32: 8 contiguous floats (2x b128) -> 8 halves (1x b128)
    {
      const float4* ap = (const float4*)(A + (size_t)(m0 + ar) * K + (kk0 + ac));
      float4 f0 = ap[0];
      float4 f1 = ap[1];
      v8h h;
      h[0] = (_Float16)f0.x; h[1] = (_Float16)f0.y;
      h[2] = (_Float16)f0.z; h[3] = (_Float16)f0.w;
      h[4] = (_Float16)f1.x; h[5] = (_Float16)f1.y;
      h[6] = (_Float16)f1.z; h[7] = (_Float16)f1.w;
      *(v8h*)&sA[bsel][ar * 32 + ac] = h;
    }
    // ---- B tile 32x64: 8 contiguous cols of one k row, stored transposed
    {
      float4 f0 = {0.f, 0.f, 0.f, 0.f};
      float4 f1 = {0.f, 0.f, 0.f, 0.f};
      const float* wp = W + (size_t)(kk0 + bk) * ldw + n0 + bc;
      if (fullN) {
        f0 = *(const float4*)(wp);
        f1 = *(const float4*)(wp + 4);
      } else {
        if (n0 + bc     < N) f0 = *(const float4*)(wp);
        if (n0 + bc + 4 < N) f1 = *(const float4*)(wp + 4);
      }
      _Float16 hh[8];
      hh[0] = (_Float16)f0.x; hh[1] = (_Float16)f0.y;
      hh[2] = (_Float16)f0.z; hh[3] = (_Float16)f0.w;
      hh[4] = (_Float16)f1.x; hh[5] = (_Float16)f1.y;
      hh[6] = (_Float16)f1.z; hh[7] = (_Float16)f1.w;
#pragma unroll
      for (int j = 0; j < 8; ++j) sBt[bsel][(bc + j) * 32 + bk] = hh[j];
    }
  };

  v8f acc0 = {}, acc1 = {};
  stage(0, 0);                    // prologue: first k-tile into buffer 0
  int buf = 0;
  for (int k0 = 0; k0 < K; k0 += 32) {
    __syncthreads();              // staged tile in `buf` is ready
    if (k0 + 32 < K) stage(buf ^ 1, k0 + 32);   // prefetch next tile

    // ---- fragments: contiguous 128-bit LDS loads.
    // A frag: row = lane&15; lanes 0-15 hold K=0..15, lanes 16-31 K=16..31
    V16 af, bf0, bf1;
    {
      const _Float16* apz = &sA[buf][(wm + lrow) * 32 + hsel * 16];
      af.h[0] = *(const v8h*)(apz);
      af.h[1] = *(const v8h*)(apz + 8);
    }
    // B frag: column = lane&15; lanes 0-15 hold K=0..15, lanes 16-31 K=16..31
    {
      const _Float16* bp0 = &sBt[buf][(wn + lrow) * 32 + hsel * 16];
      bf0.h[0] = *(const v8h*)(bp0);
      bf0.h[1] = *(const v8h*)(bp0 + 8);
      const _Float16* bp1 = &sBt[buf][(wn + 16 + lrow) * 32 + hsel * 16];
      bf1.h[0] = *(const v8h*)(bp1);
      bf1.h[1] = *(const v8h*)(bp1 + 8);
    }
    acc0 = __builtin_amdgcn_wmma_f32_16x16x32_f16(
        false, af.v, false, bf0.v, (short)0, acc0, false, false);
    acc1 = __builtin_amdgcn_wmma_f32_16x16x32_f16(
        false, af.v, false, bf1.v, (short)0, acc1, false, false);
    buf ^= 1;
  }
  // C/D layout: VGPR r -> row r (lanes 0-15) / row r+8 (lanes 16-31), col = lane&15
  const int col0 = n0 + wn + lrow;
  const int col1 = col0 + 16;
  if (col0 < N) {
    const float bv = bias ? bias[col0] : 0.0f;
#pragma unroll
    for (int r = 0; r < 8; ++r) {
      const int row = m0 + wm + hsel * 8 + r;
      float v = acc0[r] + bv;
      if (relu) v = fmaxf(v, 0.0f);
      C[(size_t)row * N + col0] = v;
    }
  }
  if (col1 < N) {
    const float bv = bias ? bias[col1] : 0.0f;
#pragma unroll
    for (int r = 0; r < 8; ++r) {
      const int row = m0 + wm + hsel * 8 + r;
      float v = acc1[r] + bv;
      if (relu) v = fmaxf(v, 0.0f);
      C[(size_t)row * N + col1] = v;
    }
  }
}

// ======================= small GEMM (K or N tiny) =======================
__global__ __launch_bounds__(256) void small_gemm_kernel(
    const float* __restrict__ A, const float* __restrict__ W,
    const float* __restrict__ bias, const float* __restrict__ rowadd, int rowmod,
    float* __restrict__ C, int M, int N, int K, int ldw, int relu)
{
  size_t i = (size_t)blockIdx.x * 256 + threadIdx.x;
  if (i >= (size_t)M * N) return;
  int col = (int)(i % N);
  size_t row = i / N;
  float s = bias ? bias[col] : 0.0f;
  const float* a = A + row * K;
  for (int k = 0; k < K; ++k) s += a[k] * W[(size_t)k * ldw + col];
  if (rowadd) s += rowadd[(row % rowmod) * N + col];
  if (relu) s = fmaxf(s, 0.0f);
  C[i] = s;
}

// ======================= anchors + memory masking =======================
__global__ __launch_bounds__(256) void anchors_mask_kernel(
    const float* __restrict__ mem, float* __restrict__ memM, float* __restrict__ anch)
{
  int l = blockIdx.x;
  int t = threadIdx.x;
  int lvl, i0, g;
  if (l < 6400)      { lvl = 0; i0 = l;        g = 80; }
  else if (l < 8000) { lvl = 1; i0 = l - 6400; g = 40; }
  else               { lvl = 2; i0 = l - 8000; g = 20; }
  int gy = i0 / g, gx = i0 % g;
  float ax = (gx + 0.5f) / g;
  float ay = (gy + 0.5f) / g;
  float aw = 0.05f * (float)(1 << lvl);
  bool valid = (ax > 0.01f) && (ax < 0.99f) && (ay > 0.01f) && (ay < 0.99f) &&
               (aw > 0.01f) && (aw < 0.99f);
  if (t < 4) {
    float a = (t == 0) ? ax : (t == 1) ? ay : aw;
    anch[(size_t)l * 4 + t] = valid ? logf(a / (1.0f - a)) : __builtin_inff();
  }
  for (int b = 0; b < CB; ++b) {
    size_t idx = ((size_t)b * CL + l) * CH + t;
    memM[idx] = valid ? mem[idx] : 0.0f;
  }
}

// ======================= fused residual + LayerNorm (H=256) =======================
__global__ __launch_bounds__(256) void add_ln_kernel(
    const float* __restrict__ x, const float* __restrict__ res,
    const float* __restrict__ s, const float* __restrict__ b,
    float* __restrict__ out)
{
  __shared__ float red[256];
  __shared__ float s_mean, s_rstd;
  size_t row = blockIdx.x;
  int t = threadIdx.x;
  float v = x[row * CH + t] + (res ? res[row * CH + t] : 0.0f);
  red[t] = v; __syncthreads();
  for (int o = 128; o > 0; o >>= 1) { if (t < o) red[t] += red[t + o]; __syncthreads(); }
  if (t == 0) s_mean = red[0] * (1.0f / CH);
  __syncthreads();
  float d = v - s_mean;
  red[t] = d * d; __syncthreads();
  for (int o = 128; o > 0; o >>= 1) { if (t < o) red[t] += red[t + o]; __syncthreads(); }
  if (t == 0) s_rstd = rsqrtf(red[0] * (1.0f / CH) + 1e-5f);
  __syncthreads();
  out[row * CH + t] = d * s_rstd * s[t] + b[t];
}

// ======================= misc elementwise =======================
__global__ __launch_bounds__(256) void add_kernel(
    const float* __restrict__ a, const float* __restrict__ b,
    float* __restrict__ c, size_t n)
{
  size_t i = (size_t)blockIdx.x * 256 + threadIdx.x;
  if (i < n) c[i] = a[i] + b[i];
}

__global__ __launch_bounds__(256) void rowmax_kernel(
    const float* __restrict__ logits, float* __restrict__ sc, int rows, int nc)
{
  int i = blockIdx.x * 256 + threadIdx.x;
  if (i >= rows) return;
  const float* p = logits + (size_t)i * nc;
  float m = p[0];
  for (int c = 1; c < nc; ++c) m = fmaxf(m, p[c]);
  sc[i] = m;
}

// ======================= deterministic iterative top-k =======================
__global__ __launch_bounds__(256) void topk_kernel(float* __restrict__ sc, int* __restrict__ idx)
{
  __shared__ float bv[256];
  __shared__ int   bi[256];
  int b = blockIdx.x;
  float* s = sc + (size_t)b * CL;
  int t = threadIdx.x;
  for (int sel = 0; sel < CNQ; ++sel) {
    float best = -__builtin_inff(); int besti = 0;
    for (int l = t; l < CL; l += 256) {
      float v = s[l];
      if (v > best) { best = v; besti = l; }
    }
    bv[t] = best; bi[t] = besti;
    __syncthreads();
    for (int o = 128; o > 0; o >>= 1) {
      if (t < o) {
        if (bv[t + o] > bv[t] || (bv[t + o] == bv[t] && bi[t + o] < bi[t])) {
          bv[t] = bv[t + o]; bi[t] = bi[t + o];
        }
      }
      __syncthreads();
    }
    if (t == 0) { idx[b * CNQ + sel] = bi[0]; s[bi[0]] = -__builtin_inff(); }
    __syncthreads();
  }
}

__global__ __launch_bounds__(256) void gather_kernel(
    const float* __restrict__ outMem, const float* __restrict__ coord,
    const int* __restrict__ idx, float* __restrict__ state, float* __restrict__ ref)
{
  int bq = blockIdx.x;
  int b = bq / CNQ;
  int t = threadIdx.x;
  int id = idx[bq];
  state[(size_t)bq * CH + t] = outMem[((size_t)b * CL + id) * CH + t];
  if (t < 4) {
    float v = coord[((size_t)b * CL + id) * 4 + t];
    ref[bq * 4 + t] = 1.0f / (1.0f + expf(-v));
  }
}

// ======================= self-attention pieces =======================
__global__ __launch_bounds__(256) void attn_score_kernel(
    const float* __restrict__ qk, float* __restrict__ att)
{
  size_t total = (size_t)CB * CNH * CNQ * CNQ;
  size_t i = (size_t)blockIdx.x * 256 + threadIdx.x;
  if (i >= total) return;
  int ki = (int)(i % CNQ); size_t r = i / CNQ;
  int qi = (int)(r % CNQ); r /= CNQ;
  int h  = (int)(r % CNH); int b = (int)(r / CNH);
  const float* qp = qk + ((size_t)b * CNQ + qi) * (2 * CH) + h * CHD;
  const float* kp = qk + ((size_t)b * CNQ + ki) * (2 * CH) + CH + h * CHD;
  float s = 0.0f;
#pragma unroll
  for (int d = 0; d < CHD; ++d) s += qp[d] * kp[d];
  att[i] = s * 0.17677669529663687f;   // 1/sqrt(32)
}

__global__ __launch_bounds__(256) void softmax_rows_kernel(
    float* __restrict__ att, int rows, int n)
{
  int row = blockIdx.x * 8 + (threadIdx.x >> 5);
  int lane = threadIdx.x & 31;
  if (row >= rows) return;
  float* p = att + (size_t)row * n;
  float mx = -__builtin_inff();
  for (int i = lane; i < n; i += 32) mx = fmaxf(mx, p[i]);
  for (int m = 16; m > 0; m >>= 1) mx = fmaxf(mx, __shfl_xor(mx, m, 32));
  float sum = 0.0f;
  for (int i = lane; i < n; i += 32) { float e = expf(p[i] - mx); p[i] = e; sum += e; }
  for (int m = 16; m > 0; m >>= 1) sum += __shfl_xor(sum, m, 32);
  float inv = 1.0f / sum;
  for (int i = lane; i < n; i += 32) p[i] *= inv;
}

__global__ __launch_bounds__(256) void attn_out_kernel(
    const float* __restrict__ att, const float* __restrict__ vb, float* __restrict__ out)
{
  size_t i = (size_t)blockIdx.x * 256 + threadIdx.x;   // over 4800*256
  int d  = (int)(i & 31);
  int h  = (int)((i >> 5) & 7);
  int bq = (int)(i >> 8);
  int b  = bq / CNQ;
  int qi = bq % CNQ;
  const float* arow = att + (((size_t)(b * CNH + h) * CNQ) + qi) * CNQ;
  float s = 0.0f;
  for (int k = 0; k < CNQ; ++k)
    s += arow[k] * vb[((size_t)b * CNQ + k) * CH + h * CHD + d];
  out[i] = s;
}

// ======================= MSDA pieces =======================
__global__ __launch_bounds__(256) void aw_softmax_kernel(
    const float* __restrict__ a, float* __restrict__ o)
{
  int i = blockIdx.x * 256 + threadIdx.x;      // (bq*8 + h) over 38400
  if (i >= CB * CNQ * CNH) return;
  int h = i & 7; int bq = i >> 3;
  const float* p = a + (size_t)bq * (CNH * CNLEV * CNPT) + h * (CNLEV * CNPT);
  float*       q = o + (size_t)bq * (CNH * CNLEV * CNPT) + h * (CNLEV * CNPT);
  float mx = -__builtin_inff();
  for (int j = 0; j < 12; ++j) mx = fmaxf(mx, p[j]);
  float e[12], s = 0.0f;
  for (int j = 0; j < 12; ++j) { e[j] = expf(p[j] - mx); s += e[j]; }
  float inv = 1.0f / s;
  for (int j = 0; j < 12; ++j) q[j] = e[j] * inv;
}

__global__ __launch_bounds__(256) void msda_sample_kernel(
    const float* __restrict__ val, const float* __restrict__ offb,
    const float* __restrict__ aws, const float* __restrict__ ref,
    float* __restrict__ out)
{
  int bq = blockIdx.x;                 // 0..4799
  int b  = bq / CNQ;
  int h  = threadIdx.x >> 5;
  int d  = threadIdx.x & 31;
  float rx = ref[bq * 4 + 0], ry = ref[bq * 4 + 1];
  float rw = ref[bq * 4 + 2], rh = ref[bq * 4 + 3];
  const float* orow = offb + (size_t)bq * (CNH * CNLEV * CNPT * 2);
  const float* arow = aws  + (size_t)bq * (CNH * CNLEV * CNPT);
  float acc = 0.0f;
  int start = 0;
#pragma unroll
  for (int lvl = 0; lvl < CNLEV; ++lvl) {
    const int g = 80 >> lvl;
#pragma unroll
    for (int p = 0; p < CNPT; ++p) {
      int j = (h * CNLEV + lvl) * CNPT + p;
      float lx = rx + orow[j * 2 + 0] * 0.25f * rw * 0.5f;
      float ly = ry + orow[j * 2 + 1] * 0.25f * rh * 0.5f;
      float a  = arow[j];
      float x = lx * g - 0.5f, y = ly * g - 0.5f;
      float x0f = floorf(x), y0f = floorf(y);
      float wx = x - x0f, wy = y - y0f;
      int x0 = (int)x0f, y0 = (int)y0f;
      float sum = 0.0f;
#pragma unroll
      for (int cy = 0; cy < 2; ++cy) {
#pragma unroll
        for (int cx = 0; cx < 2; ++cx) {
          int xi = x0 + cx, yi = y0 + cy;
          float wgt = (cx ? wx : 1.0f - wx) * (cy ? wy : 1.0f - wy);
          if (xi >= 0 && xi < g && yi >= 0 && yi < g) {
            sum += wgt * val[((size_t)b * CL + start + yi * g + xi) * CH + h * CHD + d];
          }
        }
      }
      acc += a * sum;
    }
    start += g * g;
  }
  out[(size_t)bq * CH + h * CHD + d] = acc;
}

// ======================= bbox refinement =======================
__global__ __launch_bounds__(256) void bbox_update_kernel(
    const float* __restrict__ bb4, const float* __restrict__ ref,
    float* __restrict__ out, int n)
{
  int i = blockIdx.x * 256 + threadIdx.x;
  if (i >= n) return;
  float r = fminf(fmaxf(ref[i], 0.0f), 1.0f);
  float inv = logf(fmaxf(r, 1e-5f)) - logf(fmaxf(1.0f - r, 1e-5f));
  out[i] = 1.0f / (1.0f + expf(-(bb4[i] + inv)));
}

// ======================= workspace layout =======================
constexpr size_t aln(size_t x) { return (x + 255) & ~(size_t)255; }
constexpr size_t SZ_BLH  = aln((size_t)CB * CL * CH * sizeof(float));   // 137.6 MB
constexpr size_t O_MEMM  = 0;
constexpr size_t O_OUTM  = O_MEMM + SZ_BLH;
constexpr size_t O_BUFA  = O_OUTM + SZ_BLH;   // t0 / h2 / per-layer val
constexpr size_t O_BUFB  = O_BUFA + SZ_BLH;   // h1 / enc logits / attn scores
constexpr size_t O_ANCH  = O_BUFB + SZ_BLH;
constexpr size_t O_COORD = O_ANCH  + aln((size_t)CL * 4 * sizeof(float));
constexpr size_t O_SCOR  = O_COORD + aln((size_t)CB * CL * 4 * sizeof(float));
constexpr size_t O_IDX   = O_SCOR  + aln((size_t)CB * CL * sizeof(float));
constexpr size_t O_REF   = O_IDX   + aln((size_t)CB * CNQ * sizeof(int));
constexpr size_t O_REF2  = O_REF   + aln((size_t)CB * CNQ * 4 * sizeof(float));
constexpr size_t O_STATE = O_REF2  + aln((size_t)CB * CNQ * 4 * sizeof(float));
constexpr size_t O_QPH   = O_STATE + aln((size_t)CB * CNQ * CH * sizeof(float));
constexpr size_t O_QPOS  = O_QPH   + aln((size_t)CB * CNQ * 2 * CH * sizeof(float));
constexpr size_t O_QBUF  = O_QPOS  + aln((size_t)CB * CNQ * CH * sizeof(float));
constexpr size_t O_QK    = O_QBUF  + aln((size_t)CB * CNQ * CH * sizeof(float));
constexpr size_t O_VB    = O_QK    + aln((size_t)CB * CNQ * 2 * CH * sizeof(float));
constexpr size_t O_ATTO  = O_VB    + aln((size_t)CB * CNQ * CH * sizeof(float));
constexpr size_t O_SA    = O_ATTO  + aln((size_t)CB * CNQ * CH * sizeof(float));
constexpr size_t O_OFFB  = O_SA    + aln((size_t)CB * CNQ * CH * sizeof(float));
constexpr size_t O_AWB   = O_OFFB  + aln((size_t)CB * CNQ * 192 * sizeof(float));
constexpr size_t O_AWS   = O_AWB   + aln((size_t)CB * CNQ * 96 * sizeof(float));
constexpr size_t O_SAMP  = O_AWS   + aln((size_t)CB * CNQ * 96 * sizeof(float));
constexpr size_t O_CA    = O_SAMP  + aln((size_t)CB * CNQ * CH * sizeof(float));
constexpr size_t O_FFNH  = O_CA    + aln((size_t)CB * CNQ * CH * sizeof(float));
constexpr size_t O_FFNO  = O_FFNH  + aln((size_t)CB * CNQ * CDFF * sizeof(float));
constexpr size_t O_T1    = O_FFNO  + aln((size_t)CB * CNQ * CH * sizeof(float));
constexpr size_t O_T2    = O_T1    + aln((size_t)CB * CNQ * CH * sizeof(float));
constexpr size_t O_BB4   = O_T2    + aln((size_t)CB * CNQ * CH * sizeof(float));
// total workspace need: ~660 MB

// layer param order after pytree flatten (dict keys sorted):
enum { L_AW_B = 0, L_AW_W, L_BB_B0, L_BB_B1, L_BB_B2, L_BB_W0, L_BB_W1, L_BB_W2,
       L_CAO_B, L_CAO_W, L_FFN_B1, L_FFN_B2, L_FFN_W1, L_FFN_W2,
       L_LN1_B, L_LN1_S, L_LN2_B, L_LN2_S, L_LN3_B, L_LN3_S,
       L_OFF_B, L_OFF_W, L_SAOUT_B, L_SAOUT_W, L_QKV_B, L_QKV_W,
       L_SCORE_B, L_SCORE_W, L_VAL_B, L_VAL_W };

static inline void launch_gemm(hipStream_t st, const float* A, const float* W,
                               const float* bias, float* C,
                               int M, int N, int K, int ldw, int relu)
{
  dim3 g((N + 63) / 64, M / 64);
  gemm_f16_wmma_kernel<<<g, 256, 0, st>>>(A, W, bias, C, N, K, ldw, relu);
}

extern "C" void kernel_launch(void* const* d_in, const int* in_sizes, int n_in,
                              void* d_out, int out_size, void* d_ws, size_t ws_size,
                              hipStream_t stream)
{
  (void)in_sizes; (void)n_in; (void)out_size; (void)ws_size;
  // ---- input mapping (jax pytree flatten: dicts sorted by key, lists in order) ----
  const float* memory = (const float*)d_in[0];
  auto P  = [&](int i) { return (const float*)d_in[i]; };
  auto LW = [&](int layer, int j) { return (const float*)d_in[13 + 30 * layer + j]; };
  // 1..3 enc_bbox_bs, 4..6 enc_bbox_ws, 7 enc_ln_b, 8 enc_ln_s, 9 enc_out_b,
  // 10 enc_out_w, 11 enc_score_b, 12 enc_score_w, 13.. layers, 193/194 qpos_bs, 195/196 qpos_ws

  char* ws = (char*)d_ws;
  auto F = [&](size_t o) { return (float*)(ws + o); };
  float* memM  = F(O_MEMM);  float* outM  = F(O_OUTM);
  float* bufA  = F(O_BUFA);  float* bufB  = F(O_BUFB);
  float* anch  = F(O_ANCH);  float* coord = F(O_COORD);
  float* scor  = F(O_SCOR);  int*   idx   = (int*)(ws + O_IDX);
  float* ref   = F(O_REF);   float* ref2  = F(O_REF2);
  float* state = F(O_STATE); float* qph   = F(O_QPH);
  float* qpos  = F(O_QPOS);  float* qbuf  = F(O_QBUF);
  float* qk    = F(O_QK);    float* vb    = F(O_VB);
  float* atto  = F(O_ATTO);  float* sa    = F(O_SA);
  float* offb  = F(O_OFFB);  float* awb   = F(O_AWB);
  float* aws   = F(O_AWS);   float* samp  = F(O_SAMP);
  float* ca    = F(O_CA);    float* ffnh  = F(O_FFNH);
  float* ffno  = F(O_FFNO);  float* t1    = F(O_T1);
  float* t2    = F(O_T2);    float* bb4   = F(O_BB4);

  const int MBL = CB * CL;       // 134400
  const int MQ  = CB * CNQ;      // 4800

  // =================== encoder head ===================
  anchors_mask_kernel<<<CL, 256, 0, stream>>>(memory, memM, anch);
  launch_gemm(stream, memM, P(10), P(9), bufA, MBL, CH, CH, CH, 0);           // enc_out
  add_ln_kernel<<<MBL, 256, 0, stream>>>(bufA, nullptr, P(8), P(7), outM);    // enc LN
  launch_gemm(stream, outM, P(4), P(1), bufB, MBL, CH, CH, CH, 1);            // bbox h1
  launch_gemm(stream, bufB, P(5), P(2), bufA, MBL, CH, CH, CH, 1);            // bbox h2
  {
    size_t n = (size_t)MBL * 4;
    small_gemm_kernel<<<(unsigned)((n + 255) / 256), 256, 0, stream>>>(
        bufA, P(6), P(3), anch, CL, coord, MBL, 4, CH, 4, 0);                 // coord (+anchors)
  }
  launch_gemm(stream, outM, P(12), P(11), bufB, MBL, CNC, CH, CNC, 0);        // enc logits
  rowmax_kernel<<<(MBL + 255) / 256, 256, 0, stream>>>(bufB, scor, MBL, CNC);
  topk_kernel<<<CB, 256, 0, stream>>>(scor, idx);
  gather_kernel<<<MQ, 256, 0, stream>>>(outM, coord, idx, state, ref);

  // =================== decoder layers ===================
  const float* qw0 = P(195); const float* qb0 = P(193);
  const float* qw1 = P(196); const float* qb1 = P(194);
  float* refp = ref; float* refn = ref2;
  float* att = bufB;   // (B,H,NQ,NQ) fits in 137.6MB buffer

  for (int i = 0; i < CNLAYER; ++i) {
    // qpos = MLP(ref): 4 -> 512 (relu) -> 256
    {
      size_t n = (size_t)MQ * (2 * CH);
      small_gemm_kernel<<<(unsigned)((n + 255) / 256), 256, 0, stream>>>(
          refp, qw0, qb0, nullptr, 1, qph, MQ, 2 * CH, 4, 2 * CH, 1);
    }
    launch_gemm(stream, qph, qw1, qb1, qpos, MQ, CH, 2 * CH, CH, 0);
    add_kernel<<<(MQ * CH + 255) / 256, 256, 0, stream>>>(state, qpos, qbuf, (size_t)MQ * CH);

    // self-attention: QK from q=state+qpos (cols 0..511 of qkv_w), V from state (cols 512..767)
    launch_gemm(stream, qbuf,  LW(i, L_QKV_W),          LW(i, L_QKV_B),          qk, MQ, 2 * CH, CH, 3 * CH, 0);
    launch_gemm(stream, state, LW(i, L_QKV_W) + 2 * CH, LW(i, L_QKV_B) + 2 * CH, vb, MQ, CH,     CH, 3 * CH, 0);
    {
      size_t n = (size_t)CB * CNH * CNQ * CNQ;
      attn_score_kernel<<<(unsigned)((n + 255) / 256), 256, 0, stream>>>(qk, att);
    }
    softmax_rows_kernel<<<(CB * CNH * CNQ + 7) / 8, 256, 0, stream>>>(att, CB * CNH * CNQ, CNQ);
    attn_out_kernel<<<(MQ * CH + 255) / 256, 256, 0, stream>>>(att, vb, atto);
    launch_gemm(stream, atto, LW(i, L_SAOUT_W), LW(i, L_SAOUT_B), sa, MQ, CH, CH, CH, 0);
    add_ln_kernel<<<MQ, 256, 0, stream>>>(state, sa, LW(i, L_LN1_S), LW(i, L_LN1_B), state);

    // deformable cross-attention
    add_kernel<<<(MQ * CH + 255) / 256, 256, 0, stream>>>(state, qpos, qbuf, (size_t)MQ * CH);
    launch_gemm(stream, qbuf, LW(i, L_OFF_W), LW(i, L_OFF_B), offb, MQ, 192, CH, 192, 0);
    launch_gemm(stream, qbuf, LW(i, L_AW_W),  LW(i, L_AW_B),  awb,  MQ, 96,  CH, 96,  0);
    aw_softmax_kernel<<<(CB * CNQ * CNH + 255) / 256, 256, 0, stream>>>(awb, aws);
    launch_gemm(stream, memM, LW(i, L_VAL_W), LW(i, L_VAL_B), bufA, MBL, CH, CH, CH, 0); // value proj
    msda_sample_kernel<<<MQ, 256, 0, stream>>>(bufA, offb, aws, refp, samp);
    launch_gemm(stream, samp, LW(i, L_CAO_W), LW(i, L_CAO_B), ca, MQ, CH, CH, CH, 0);
    add_ln_kernel<<<MQ, 256, 0, stream>>>(state, ca, LW(i, L_LN2_S), LW(i, L_LN2_B), state);

    // FFN
    launch_gemm(stream, state, LW(i, L_FFN_W1), LW(i, L_FFN_B1), ffnh, MQ, CDFF, CH,   CDFF, 1);
    launch_gemm(stream, ffnh,  LW(i, L_FFN_W2), LW(i, L_FFN_B2), ffno, MQ, CH,   CDFF, CH,   0);
    add_ln_kernel<<<MQ, 256, 0, stream>>>(state, ffno, LW(i, L_LN3_S), LW(i, L_LN3_B), state);

    // bbox refinement
    launch_gemm(stream, state, LW(i, L_BB_W0), LW(i, L_BB_B0), t1, MQ, CH, CH, CH, 1);
    launch_gemm(stream, t1,    LW(i, L_BB_W1), LW(i, L_BB_B1), t2, MQ, CH, CH, CH, 1);
    {
      size_t n = (size_t)MQ * 4;
      small_gemm_kernel<<<(unsigned)((n + 255) / 256), 256, 0, stream>>>(
          t2, LW(i, L_BB_W2), LW(i, L_BB_B2), nullptr, 1, bb4, MQ, 4, CH, 4, 0);
    }
    float* dstref = (i == CNLAYER - 1) ? ((float*)d_out + (size_t)MQ * CNC) : refn;
    bbox_update_kernel<<<(MQ * 4 + 255) / 256, 256, 0, stream>>>(bb4, refp, dstref, MQ * 4);

    if (i == CNLAYER - 1) {
      // final class logits straight into d_out (compact (B*NQ, 80) row-major)
      launch_gemm(stream, state, LW(i, L_SCORE_W), LW(i, L_SCORE_B), (float*)d_out, MQ, CNC, CH, CNC, 0);
    } else {
      float* tmp = refp; refp = refn; refn = tmp;
    }
  }
}
